// PrototypeAgentNetwork_8615704396241
// MI455X (gfx1250) — matrix operand order repaired and checked by reading, not compiled, not run on hardware
//
#include <hip/hip_runtime.h>
#include <cmath>
#include <cstdint>

// ---------------- Types for CDNA5 WMMA ----------------
typedef _Float16 h16;
typedef __attribute__((ext_vector_type(16))) _Float16 v16h;
typedef __attribute__((ext_vector_type(8)))  float    v8f;

// ---------------- Problem constants (from reference) ----------------
#define CC     256     // H_DIM
#define CIN    512     // IN_DIM
#define HWC    4096    // 64*64
#define NB     4       // batch
#define QNF    4       // q frames per batch elem
#define NFR    16      // total frames (q or s)
#define TQ     16384   // tokens per batch elem (QNF*HWC)
#define KQ     20      // PPF * QNF query protos
#define KS     5       // PPF support protos per frame
#define KMIT   20
#define LAMS   0.8f
#define LAMC   0.2f

// ================= WMMA GEMM: C[M,N] = A[M,K](f16) * B[K,N](f16, swizzled) + bias ==========
#define TM 128
#define TN 64
#define TK 32

// CDNA5 async tensor path: DMA memory -> LDS without VGPR staging (ASYNCcnt tracked).
__device__ __forceinline__ void async_copy_b128(uint32_t lds_addr, uint64_t gaddr) {
  asm volatile("global_load_async_to_lds_b128 %0, %1, off"
               :: "v"(lds_addr), "v"(gaddr)
               : "memory");
}

__device__ __forceinline__ v16h frag_a(const h16* __restrict__ ls, int r0, int lane) {
  // 16-bit A 16x32 layout: lane half selects K offset 0/16,
  // VGPR v: pair = v*2 (contiguous K, matching frag_b).
  // Per lane this is one contiguous 16-element (32B) run.
  v16h a;
  const h16* row = ls + (r0 + (lane & 15)) * TK;
  int koff = (lane >> 4) << 4;
#pragma unroll
  for (int v = 0; v < 8; ++v) {
    int kb = koff + (v << 1);
    a[2 * v]     = row[kb];
    a[2 * v + 1] = row[kb + 1];
  }
  return a;
}

__device__ __forceinline__ v16h frag_b(const h16* __restrict__ ls, int c0, int lane) {
  // B tile stored column-major per 32-K block: ls[n*TK + k].
  // Lane fragment = K = kh..kh+15 at fixed column n -> 32 contiguous bytes.
  const h16* colp = ls + (c0 + (lane & 15)) * TK + ((lane >> 4) << 4);
  return *(const v16h*)colp;
}

__global__ __launch_bounds__(256) void k_gemm_f16(
    const h16* __restrict__ A, const h16* __restrict__ Bm,
    const float* __restrict__ bias, float* __restrict__ Cout,
    int M, int N, int K)
{
  __shared__ __align__(32) h16 lsA[2][TM * TK];
  __shared__ __align__(32) h16 lsB[2][TN * TK];   // column-major within K-block
  const int tid  = threadIdx.x;
  const int lane = tid & 31;
  const int wave = tid >> 5;     // 0..7
  const int wm   = wave & 3;     // 4 row groups of 32
  const int wn   = wave >> 2;    // 2 col groups of 32
  const int m0   = blockIdx.y * TM;
  const int n0   = blockIdx.x * TN;

  __builtin_prefetch(bias + n0, 0, 0);   // global_prefetch_b8: epilogue bias row

  // per-thread async tile-copy addressing (3 b128 DMAs per thread per stage)
  const int rA  = tid >> 2;
  const int cA8 = (tid & 3) << 3;
  const int cB  = tid >> 2;          // 0..63: column of B tile
  const int hB8 = (tid & 3) << 3;    // 0,8,16,24 within 32-elem column

  auto issue_stage = [&](int bufi, int k0) {
    async_copy_b128((uint32_t)(uintptr_t)(&lsA[bufi][rA * TK + cA8]),
                    (uint64_t)(uintptr_t)(A + (size_t)(m0 + rA) * K + k0 + cA8));
    async_copy_b128((uint32_t)(uintptr_t)(&lsA[bufi][(rA + 64) * TK + cA8]),
                    (uint64_t)(uintptr_t)(A + (size_t)(m0 + rA + 64) * K + k0 + cA8));
    async_copy_b128((uint32_t)(uintptr_t)(&lsB[bufi][cB * TK + hB8]),
                    (uint64_t)(uintptr_t)(Bm + ((size_t)(k0 >> 5) * N + n0 + cB) * TK + hB8));
  };

  v8f acc[2][2];
#pragma unroll
  for (int i = 0; i < 2; ++i)
#pragma unroll
    for (int j = 0; j < 2; ++j)
#pragma unroll
      for (int v = 0; v < 8; ++v) acc[i][j][v] = 0.0f;

  const int nk = K / TK;
  issue_stage(0, 0);                      // prologue: DMA tile 0
  for (int kt = 0; kt < nk; ++kt) {
    const int buf = kt & 1;
    if (kt + 1 < nk) {
      issue_stage(buf ^ 1, (kt + 1) * TK);              // overlap DMA of next tile
      asm volatile("s_wait_asynccnt 0x3" ::: "memory"); // this wave's current tile landed
    } else {
      asm volatile("s_wait_asynccnt 0x0" ::: "memory");
    }
    __syncthreads();                       // cross-wave visibility of LDS tile
    v16h af[2], bf[2];
#pragma unroll
    for (int i = 0; i < 2; ++i) af[i] = frag_a(lsA[buf], wm * 32 + i * 16, lane);
#pragma unroll
    for (int j = 0; j < 2; ++j) bf[j] = frag_b(lsB[buf], wn * 32 + j * 16, lane);
#pragma unroll
    for (int i = 0; i < 2; ++i)
#pragma unroll
      for (int j = 0; j < 2; ++j)
        acc[i][j] = __builtin_amdgcn_wmma_f32_16x16x32_f16(
            false, af[i], false, bf[j], (short)0, acc[i][j], false, false);
    __syncthreads();                       // protect buf before stage kt+2 DMA targets it
  }
  // C/D layout: VGPR v -> row r0 + v + (lane>>4)*8, col c0 + (lane&15)
#pragma unroll
  for (int i = 0; i < 2; ++i)
#pragma unroll
    for (int j = 0; j < 2; ++j) {
      int col   = n0 + wn * 32 + j * 16 + (lane & 15);
      int rbase = m0 + wm * 32 + i * 16 + ((lane >> 4) << 3);
      float bv  = bias[col];
#pragma unroll
      for (int v = 0; v < 8; ++v)
        Cout[(size_t)(rbase + v) * N + col] = acc[i][j][v] + bv;
    }
}

// ================= Prep / layout kernels =================
__global__ void k_wt(const float* __restrict__ W, h16* __restrict__ WT, int N, int K) {
  // fragment-ready swizzle: WT[((k/32)*N + n)*32 + (k%32)] = W[n*K + k]
  int i = blockIdx.x * 256 + threadIdx.x;
  if (i >= K * N) return;
  int k = i / N, n = i % N;
  int kb = k >> 5, kk = k & 31;
  WT[((size_t)kb * N + n) * 32 + kk] = (h16)W[(size_t)n * K + k];
}

__global__ void k_prep(const float* __restrict__ feat, const float* __restrict__ mask,
                       h16* __restrict__ A) {
  // A[(n*HWC+p)*CIN + c] = feat[(n*CIN+c)*HWC + p] * mask[n*HWC+p]
  size_t i = (size_t)blockIdx.x * 256 + threadIdx.x;
  int c = (int)(i % CIN);
  size_t np = i / CIN;
  int p = (int)(np % HWC);
  int n = (int)(np / HWC);
  float v = feat[((size_t)n * CIN + c) * HWC + p] * mask[(size_t)n * HWC + p];
  A[i] = (h16)v;
}

__global__ void k_cvt(const float* __restrict__ in, h16* __restrict__ out) {
  size_t i = (size_t)blockIdx.x * 256 + threadIdx.x;
  out[i] = (h16)in[i];
}

// ================= K-means =================
__global__ __launch_bounds__(256) void k_topk_init(
    const float* __restrict__ mask, const float* __restrict__ data,
    float* __restrict__ cent, float* __restrict__ thr, int P, int K)
{
  int g = blockIdx.x;
  const float* m = mask + (size_t)g * P;
  const float* d = data + (size_t)g * P * CC;
  float* cg = cent + (size_t)g * K * CC;
  __shared__ float red[256];
  __shared__ int   redi[256];
  __shared__ int   sel[32];
  int tid = threadIdx.x;
  // mean -> threshold
  float s = 0.f;
  for (int p = tid; p < P; p += 256) s += m[p];
  red[tid] = s; __syncthreads();
  for (int off = 128; off > 0; off >>= 1) {
    if (tid < off) red[tid] += red[tid + off];
    __syncthreads();
  }
  if (tid == 0) thr[g] = red[0] / (float)P;
  __syncthreads();
  // stable top-K of mask (descending, ties -> smaller index)
  for (int kk = 0; kk < K; ++kk) {
    float bv = -1e30f; int bi = 0x7fffffff;
    for (int p = tid; p < P; p += 256) {
      bool taken = false;
      for (int t = 0; t < kk; ++t) if (sel[t] == p) taken = true;
      if (taken) continue;
      float v = m[p];
      if (v > bv || (v == bv && p < bi)) { bv = v; bi = p; }
    }
    red[tid] = bv; redi[tid] = bi; __syncthreads();
    for (int off = 128; off > 0; off >>= 1) {
      if (tid < off) {
        float v2 = red[tid + off]; int i2 = redi[tid + off];
        if (v2 > red[tid] || (v2 == red[tid] && i2 < redi[tid])) { red[tid] = v2; redi[tid] = i2; }
      }
      __syncthreads();
    }
    if (tid == 0) sel[kk] = redi[0];
    __syncthreads();
  }
  for (int e = tid; e < K * CC; e += 256)
    cg[e] = d[(size_t)sel[e / CC] * CC + (e % CC)];
}

template <int K>
__global__ __launch_bounds__(256) void k_assign_pass(
    const float* __restrict__ data, const float* __restrict__ mask, const float* __restrict__ thr,
    const float* __restrict__ cent, int* __restrict__ asg, int P)
{
  int g = blockIdx.y;
  __shared__ float sc[K * CC];
  int tid = threadIdx.x;
  const float* cg = cent + (size_t)g * K * CC;
  for (int e = tid; e < K * CC; e += 256) sc[e] = cg[e];
  __syncthreads();
  int p = blockIdx.x * 256 + tid;
  if (p >= P) return;
  const float* dr = data + ((size_t)g * P + p) * CC;
  float dots[K];
#pragma unroll
  for (int kk = 0; kk < K; ++kk) dots[kk] = 0.f;
  for (int c = 0; c < CC; ++c) {
    float dv = dr[c];
#pragma unroll
    for (int kk = 0; kk < K; ++kk) dots[kk] += dv * sc[kk * CC + c];
  }
  float best = dots[0]; int bk = 0;
#pragma unroll
  for (int kk = 1; kk < K; ++kk) if (dots[kk] > best) { best = dots[kk]; bk = kk; }
  bool valid = mask[(size_t)g * P + p] >= thr[g];
  asg[(size_t)g * P + p] = valid ? bk : -1;
}

__global__ __launch_bounds__(256) void k_cent_update(
    const float* __restrict__ data, const int* __restrict__ asg,
    float* __restrict__ cent, int P, int K)
{
  int gk = blockIdx.x, g = gk / K, k = gk % K;
  int c = threadIdx.x;
  const float* dg = data + (size_t)g * P * CC;
  const int*   ag = asg + (size_t)g * P;
  float acc = 0.f, cnt = 0.f;
  for (int p = 0; p < P; ++p)
    if (ag[p] == k) { acc += dg[(size_t)p * CC + c]; cnt += 1.f; }
  if (cnt > 0.f) cent[(size_t)gk * CC + c] = acc / cnt;  // counts>0 -> /counts; else keep old
}

// ================= Small linears & graph attention =================
__global__ __launch_bounds__(256) void k_lin_small(
    const float* __restrict__ X, const float* __restrict__ W, const float* __restrict__ b,
    float* __restrict__ Y, int T)
{
  int t = blockIdx.x, g = blockIdx.y, tid = threadIdx.x;
  __shared__ float xs[CC];
  const float* xr = X + ((size_t)g * T + t) * CC;
  xs[tid] = xr[tid];
  __syncthreads();
  float acc = b[tid];
  const float* wr = W + (size_t)tid * CC;
  for (int c = 0; c < CC; ++c) acc += xs[c] * wr[c];
  Y[((size_t)g * T + t) * CC + tid] = acc;
}

__global__ __launch_bounds__(256) void k_graph(
    const float* __restrict__ qp, const float* __restrict__ kp, const float* __restrict__ vp,
    const float* __restrict__ base, float* __restrict__ out, float lam)
{
  const int T = KQ;
  int g = blockIdx.x, tid = threadIdx.x;
  __shared__ float qn[KQ], kn[KQ], cosm[KQ * KQ], rs[KQ];
  const float* qg = qp + (size_t)g * T * CC;
  const float* kg = kp + (size_t)g * T * CC;
  const float* vg = vp + (size_t)g * T * CC;
  if (tid < 2 * T) {
    const float* r = (tid < T) ? (qg + (size_t)tid * CC) : (kg + (size_t)(tid - T) * CC);
    float s = 0.f;
    for (int c = 0; c < CC; ++c) s += r[c] * r[c];
    if (tid < T) qn[tid] = sqrtf(s); else kn[tid - T] = sqrtf(s);
  }
  __syncthreads();
  for (int e = tid; e < T * T; e += 256) {
    int t = e / T, j = e % T;
    float d = 0.f;
    for (int c = 0; c < CC; ++c) d += qg[(size_t)t * CC + c] * kg[(size_t)j * CC + c];
    cosm[e] = d / (qn[t] * kn[j] + 1e-8f);
  }
  __syncthreads();
  if (tid < T) {
    float s = 0.f;
    for (int j = 0; j < T; ++j) s += cosm[tid * T + j];
    rs[tid] = s;
  }
  __syncthreads();
  for (int e = tid; e < T * CC; e += 256) {
    int t = e / CC, c = e % CC;
    float a = 0.f;
    for (int j = 0; j < T; ++j) a += (cosm[t * T + j] / rs[t]) * vg[(size_t)j * CC + c];
    out[(size_t)g * T * CC + e] = base[(size_t)g * T * CC + e] + lam * a;
  }
}

// ================= Agent attention pieces =================
__global__ __launch_bounds__(256) void k_attn1(
    const float* __restrict__ aq, const float* __restrict__ keys,
    float* __restrict__ qk, int T2, float inv_scale)
{
  const int T1 = KQ;
  int g = blockIdx.y;
  int j = blockIdx.x * 256 + threadIdx.x;
  __shared__ float aqs[KQ * CC];
  for (int e = threadIdx.x; e < T1 * CC; e += 256) aqs[e] = aq[(size_t)g * T1 * CC + e];
  __syncthreads();
  const float* kr = keys + ((size_t)g * T2 + j) * CC;
  for (int i = 0; i < T1; ++i) {
    float d = 0.f;
    for (int c = 0; c < CC; ++c) d += aqs[i * CC + c] * kr[c];
    qk[((size_t)g * T1 + i) * T2 + j] = d * inv_scale;
  }
}

__global__ __launch_bounds__(256) void k_sm_ms(
    const float* __restrict__ qk, float* __restrict__ mrow, float* __restrict__ srow, int T2)
{
  int gi = blockIdx.x, tid = threadIdx.x;
  const float* r = qk + (size_t)gi * T2;
  __shared__ float red[256];
  float m = -1e30f;
  for (int j = tid; j < T2; j += 256) m = fmaxf(m, r[j]);
  red[tid] = m; __syncthreads();
  for (int off = 128; off > 0; off >>= 1) {
    if (tid < off) red[tid] = fmaxf(red[tid], red[tid + off]);
    __syncthreads();
  }
  float mv = red[0];
  __syncthreads();
  float s = 0.f;
  for (int j = tid; j < T2; j += 256) s += expf(r[j] - mv);
  red[tid] = s; __syncthreads();
  for (int off = 128; off > 0; off >>= 1) {
    if (tid < off) red[tid] += red[tid + off];
    __syncthreads();
  }
  if (tid == 0) { mrow[gi] = mv; srow[gi] = red[0]; }
}

__global__ __launch_bounds__(256) void k_astok(
    const float* __restrict__ qk, const float* __restrict__ mrow, const float* __restrict__ srow,
    const float* __restrict__ vmat, const float* __restrict__ atok,
    float* __restrict__ out, int T2)
{
  const int T1 = KQ;
  int gi = blockIdx.x;            // g*T1 + i
  int g = gi / T1;
  int c = threadIdx.x;
  const float* r = qk + (size_t)gi * T2;
  float mv = mrow[gi], inv = 1.f / srow[gi];
  float acc = 0.f;
  const float* vb = vmat + (size_t)g * T2 * CC + c;
  for (int j = 0; j < T2; ++j) acc += expf(r[j] - mv) * vb[(size_t)j * CC];
  out[(size_t)gi * CC + c] = atok[(size_t)gi * CC + c] + acc * inv;
}

__global__ __launch_bounds__(256) void k_attn2(
    const float* __restrict__ qlin, const float* __restrict__ ak,
    const float* __restrict__ astok, const float* __restrict__ qtok,
    float* __restrict__ outp, int T, int coff, float inv_scale)
{
  const int T1 = KQ;
  int g = blockIdx.y, t = blockIdx.x, tid = threadIdx.x;
  __shared__ float qs_[CC];
  __shared__ float wts[KQ];
  qs_[tid] = qlin[((size_t)g * T + t) * CC + tid];
  __syncthreads();
  if (tid < T1) {
    const float* kr = ak + ((size_t)g * T1 + tid) * CC;
    float d = 0.f;
    for (int c = 0; c < CC; ++c) d += qs_[c] * kr[c];
    wts[tid] = d * inv_scale;
  }
  __syncthreads();
  if (tid == 0) {
    float mv = -1e30f;
    for (int i = 0; i < T1; ++i) mv = fmaxf(mv, wts[i]);
    float s = 0.f;
    for (int i = 0; i < T1; ++i) { wts[i] = expf(wts[i] - mv); s += wts[i]; }
    float inv = 1.f / s;
    for (int i = 0; i < T1; ++i) wts[i] *= inv;
  }
  __syncthreads();
  float acc = qtok[((size_t)g * T + t) * CC + tid];
  const float* ab = astok + (size_t)g * T1 * CC + tid;
  for (int i = 0; i < T1; ++i) acc += wts[i] * ab[(size_t)i * CC];
  int frame = g * QNF + (t / HWC);
  int p = t % HWC;
  outp[((size_t)frame * (2 * CC) + coff + tid) * HWC + p] = acc;
}

// ================= Host orchestration =================
extern "C" void kernel_launch(void* const* d_in, const int* in_sizes, int n_in,
                              void* d_out, int out_size, void* d_ws, size_t ws_size,
                              hipStream_t stream) {
  (void)in_sizes; (void)n_in; (void)out_size; (void)ws_size;
  const float* q_feat = (const float*)d_in[0];
  const float* s_feat = (const float*)d_in[1];
  const float* q_mask = (const float*)d_in[2];
  const float* s_mask = (const float*)d_in[3];
  const float* conv_q_w = (const float*)d_in[4];
  const float* conv_q_b = (const float*)d_in[5];
  const float* conv_s_w = (const float*)d_in[6];
  const float* conv_s_b = (const float*)d_in[7];
  struct AttnP { const float *qw,*qb,*kw,*kb,*vw,*vb; };
  auto getp = [&](int base) {
    return AttnP{ (const float*)d_in[base+0], (const float*)d_in[base+1],
                  (const float*)d_in[base+2], (const float*)d_in[base+3],
                  (const float*)d_in[base+4], (const float*)d_in[base+5] };
  };
  AttnP Pqp = getp(8), Psp = getp(14), Pco = getp(20), Pqs = getp(26), Pqq = getp(32);
  float* dout = (float*)d_out;

  // ---- workspace carve (bump allocator, 256B aligned) ----
  char* ws = (char*)d_ws;
  size_t off = 0;
  auto alloc = [&](size_t bytes) -> void* {
    void* p = ws + off;
    off = (off + bytes + 255) & ~(size_t)255;
    return p;
  };
  const size_t FT_F32 = (size_t)NFR * HWC * CC * 4;   // 64MB
  const size_t FT_F16 = (size_t)NFR * HWC * CC * 2;   // 32MB
  const size_t ST_SZ  = (size_t)NFR * HWC * CIN * 2;  // 64MB (== lin f32 size)
  float* QF   = (float*)alloc(FT_F32);
  float* SF   = (float*)alloc(FT_F32);
  h16*   QB   = (h16*)  alloc(FT_F16);
  h16*   SB   = (h16*)  alloc(FT_F16);
  void*  STv  =         alloc(ST_SZ);     // f16 staging, later LIN1 (f32)
  float* LIN2 = (float*)alloc(ST_SZ);
  float* LIN3 = (float*)alloc(ST_SZ);
  float* QK   = (float*)alloc((size_t)NB * KQ * TQ * 4);
  h16* WqT = (h16*)alloc((size_t)CIN * CC * 2);
  h16* WsT = (h16*)alloc((size_t)CIN * CC * 2);
  h16* WT6[6];
  for (int i = 0; i < 6; ++i) WT6[i] = (h16*)alloc((size_t)CC * CC * 2);
  float* thrQ = (float*)alloc(NB * 4);
  float* thrS = (float*)alloc(NFR * 4);
  const size_t PROTO = (size_t)NB * KQ * CC * 4;      // 80KB
  float* q_proto  = (float*)alloc(PROTO);
  float* s_proto  = (float*)alloc(PROTO);             // [16][5][256] == [4][20][256]
  int*   asgQ     = (int*)  alloc((size_t)NB * TQ * 4);
  int*   asgS     = (int*)  alloc((size_t)NFR * HWC * 4);
  float* gq = (float*)alloc(PROTO);
  float* gk = (float*)alloc(PROTO);
  float* gv = (float*)alloc(PROTO);
  float* qq_proto = (float*)alloc(PROTO);
  float* ss_proto = (float*)alloc(PROTO);
  float* a_tok    = (float*)alloc(PROTO);
  float* aqb      = (float*)alloc(PROTO);
  float* akb      = (float*)alloc(PROTO);
  float* as_tok   = (float*)alloc(PROTO);
  float* mrow = (float*)alloc(NB * KQ * 4);
  float* srow = (float*)alloc(NB * KQ * 4);

  h16* ST = (h16*)STv;       // conv staging (f16)
  float* LIN1 = (float*)STv; // reused after conv phase (f32)

  const int M_TOK = NFR * HWC;  // 65536

  // ---- weight transposes (f32 -> f16, fragment-ready swizzle) ----
  k_wt<<<(CIN * CC + 255) / 256, 256, 0, stream>>>(conv_q_w, WqT, CC, CIN);
  k_wt<<<(CIN * CC + 255) / 256, 256, 0, stream>>>(conv_s_w, WsT, CC, CIN);
  const float* bigW[6] = { Pqs.qw, Pqs.kw, Pqs.vw, Pqq.qw, Pqq.kw, Pqq.vw };
  for (int i = 0; i < 6; ++i)
    k_wt<<<(CC * CC + 255) / 256, 256, 0, stream>>>(bigW[i], WT6[i], CC, CC);

  // ---- masked 1x1 convs via WMMA GEMM (async double-buffered tiles) ----
  const size_t prepTot = (size_t)NFR * HWC * CIN;
  k_prep<<<(unsigned)(prepTot / 256), 256, 0, stream>>>(q_feat, q_mask, ST);
  k_gemm_f16<<<dim3(CC / TN, M_TOK / TM), 256, 0, stream>>>(ST, WqT, conv_q_b, QF, M_TOK, CC, CIN);
  k_prep<<<(unsigned)(prepTot / 256), 256, 0, stream>>>(s_feat, s_mask, ST);
  k_gemm_f16<<<dim3(CC / TN, M_TOK / TM), 256, 0, stream>>>(ST, WsT, conv_s_b, SF, M_TOK, CC, CIN);

  // f16 copies of token features for later WMMA linears
  const size_t cvtTot = (size_t)M_TOK * CC;
  k_cvt<<<(unsigned)(cvtTot / 256), 256, 0, stream>>>(QF, QB);
  k_cvt<<<(unsigned)(cvtTot / 256), 256, 0, stream>>>(SF, SB);

  // ---- k-means prototypes (deterministic two-pass) ----
  k_topk_init<<<NB, 256, 0, stream>>>(q_mask, QF, q_proto, thrQ, TQ, KQ);
  for (int it = 0; it < KMIT; ++it) {
    k_assign_pass<KQ><<<dim3(TQ / 256, NB), 256, 0, stream>>>(QF, q_mask, thrQ, q_proto, asgQ, TQ);
    k_cent_update<<<NB * KQ, 256, 0, stream>>>(QF, asgQ, q_proto, TQ, KQ);
  }
  k_topk_init<<<NFR, 256, 0, stream>>>(s_mask, SF, s_proto, thrS, HWC, KS);
  for (int it = 0; it < KMIT; ++it) {
    k_assign_pass<KS><<<dim3(HWC / 256, NFR), 256, 0, stream>>>(SF, s_mask, thrS, s_proto, asgS, HWC);
    k_cent_update<<<NFR * KS, 256, 0, stream>>>(SF, asgS, s_proto, HWC, KS);
  }

  // ---- graph attention on prototypes ----
  auto graph = [&](const AttnP& P, const float* qi, const float* ki, const float* vi,
                   const float* base, float lam, float* out) {
    k_lin_small<<<dim3(KQ, NB), 256, 0, stream>>>(qi, P.qw, P.qb, gq, KQ);
    k_lin_small<<<dim3(KQ, NB), 256, 0, stream>>>(ki, P.kw, P.kb, gk, KQ);
    k_lin_small<<<dim3(KQ, NB), 256, 0, stream>>>(vi, P.vw, P.vb, gv, KQ);
    k_graph<<<NB, 256, 0, stream>>>(gq, gk, gv, base, out, lam);
  };
  graph(Pqp, q_proto, q_proto, q_proto, q_proto, LAMS, qq_proto);
  graph(Psp, s_proto, s_proto, s_proto, s_proto, LAMS, ss_proto);
  graph(Pco, ss_proto, qq_proto, qq_proto, ss_proto, LAMC, a_tok);

  // ---- agent attention (qs then qq) ----
  const float inv_s1 = 1.0f / sqrtf((float)KQ);   // attention(aq, sk, sv): q.shape[1] = 20
  const float inv_s2 = 1.0f / sqrtf((float)TQ);   // attention(qq, ak, as): q.shape[1] = 16384
  auto agent = [&](const AttnP& P, const h16* WTq, const h16* WTk, const h16* WTv,
                   const h16* stokB, int coff) {
    k_gemm_f16<<<dim3(CC / TN, M_TOK / TM), 256, 0, stream>>>(stokB, WTk, P.kb, LIN1, M_TOK, CC, CC);
    k_gemm_f16<<<dim3(CC / TN, M_TOK / TM), 256, 0, stream>>>(stokB, WTv, P.vb, LIN2, M_TOK, CC, CC);
    k_gemm_f16<<<dim3(CC / TN, M_TOK / TM), 256, 0, stream>>>(QB,    WTq, P.qb, LIN3, M_TOK, CC, CC);
    k_lin_small<<<dim3(KQ, NB), 256, 0, stream>>>(a_tok, P.qw, P.qb, aqb, KQ);
    k_lin_small<<<dim3(KQ, NB), 256, 0, stream>>>(a_tok, P.kw, P.kb, akb, KQ);
    k_attn1<<<dim3(TQ / 256, NB), 256, 0, stream>>>(aqb, LIN1, QK, TQ, inv_s1);
    k_sm_ms<<<NB * KQ, 256, 0, stream>>>(QK, mrow, srow, TQ);
    k_astok<<<NB * KQ, 256, 0, stream>>>(QK, mrow, srow, LIN2, a_tok, as_tok, TQ);
    k_attn2<<<dim3(TQ, NB), 256, 0, stream>>>(LIN3, akb, as_tok, QF, dout, TQ, coff, inv_s2);
  };
  agent(Pqs, WT6[0], WT6[1], WT6[2], SB, 0);    // qs: keys/values from s_tok
  agent(Pqq, WT6[3], WT6[4], WT6[5], QB, CC);   // qq: keys/values from q_tok
}
